// KANLinear_29712583754328
// MI455X (gfx1250) — matrix operand order repaired and checked by reading, not compiled, not run on hardware
//
#include <hip/hip_runtime.h>
#include <stdint.h>
#include <stddef.h>

// ---------------------------------------------------------------------------
// KANLinear fused as a single bf16 WMMA GEMM:  out[4096,1024] = A[4096,9216] x
// B[9216,1024] + bias, where A = [silu(x) | bspline_basis(x)] and
// B = [base_weight^T ; spline_weight (o,i,b) as columns].
//
// Roofline: 77.3 GFLOP vs ~68 MB of unique data -> compute bound on WMMA;
// A (75.5 MB bf16) + packed B (18.9 MB) both stay resident in the 192 MB L2.
// ---------------------------------------------------------------------------

typedef __bf16 bf16_t;
typedef bf16_t v16bf __attribute__((ext_vector_type(16)));
typedef float  v8f   __attribute__((ext_vector_type(8)));

#define N_ROWS 4096
#define IN_F   1024
#define OUT_F  1024
#define NB     8
#define KDIM   (IN_F + IN_F * NB)      // 9216
#define KTILES (KDIM / 32)             // 288
#define NTILES (OUT_F / 16)            // 64 B-fragment columns

// f32 -> bf16 round-to-nearest-even
__device__ __forceinline__ unsigned short f2bf(float f) {
    unsigned u = __float_as_uint(f);
    u += 0x7fffu + ((u >> 16) & 1u);
    return (unsigned short)(u >> 16);
}
__device__ __forceinline__ unsigned pack2(float lo, float hi) {
    return (unsigned)f2bf(lo) | ((unsigned)f2bf(hi) << 16);
}

// ---------------------------------------------------------------------------
// Kernel 1: build A (bf16, row-major [4096 x 9216]).
// One thread per (n, i): silu(x) -> col i, 8 basis values -> cols 1024+8i..+7.
// ---------------------------------------------------------------------------
__global__ void __launch_bounds__(256) prep_a(const float* __restrict__ x,
                                              unsigned short* __restrict__ A) {
    int t = blockIdx.x * blockDim.x + threadIdx.x;   // [0, 4096*1024)
    int n = t >> 10;
    int i = t & (IN_F - 1);
    float xv = x[t];

    // SiLU
    float s = xv / (1.0f + __expf(-xv));
    A[(size_t)n * KDIM + i] = f2bf(s);

    // Cox-de Boor, order 3, uniform knots t_j = -1 + (j-3)*0.4, j=0..11
    const float h = 0.4f;
    float knot[12];
#pragma unroll
    for (int j = 0; j < 12; ++j) knot[j] = -1.0f + (float)(j - 3) * h;

    float b[11];
#pragma unroll
    for (int j = 0; j < 11; ++j)
        b[j] = (xv >= knot[j] && xv < knot[j + 1]) ? 1.0f : 0.0f;

#pragma unroll
    for (int k = 1; k <= 3; ++k) {
#pragma unroll
        for (int j = 0; j < 11 - k; ++j) {
            float left  = (xv - knot[j]) / (knot[j + k] - knot[j] + 1e-8f) * b[j];
            float right = (knot[j + k + 1] - xv) /
                          (knot[j + k + 1] - knot[j + 1] + 1e-8f) * b[j + 1];
            b[j] = left + right;
        }
    }

    uint4 pk;
    pk.x = pack2(b[0], b[1]);
    pk.y = pack2(b[2], b[3]);
    pk.z = pack2(b[4], b[5]);
    pk.w = pack2(b[6], b[7]);
    *reinterpret_cast<uint4*>(A + (size_t)n * KDIM + IN_F + (size_t)i * NB) = pk;
}

// ---------------------------------------------------------------------------
// Kernel 2: build B in WMMA B-fragment order (bf16 packed in dwords).
// Fragment (kt, nt) covers K = kt*32..+31, N = nt*16..+15.  Per ISA layout:
//   lane<16 : N = lane,     dword d holds (K=(d>>2)*16+2*(d&3),   +1)
//   lane>=16: N = lane-16,  dword d holds (K=8+(d>>2)*16+2*(d&3), +1)
// Stored: Bp[((kt*64 + nt)*32 + lane)*8 + d]  (per-lane data contiguous ->
// GEMM loads each fragment as two global_load_b128 per lane).
// ---------------------------------------------------------------------------
__global__ void __launch_bounds__(256) prep_b(const float* __restrict__ base_w,
                                              const float* __restrict__ spline_w,
                                              unsigned* __restrict__ Bp) {
    int t    = blockIdx.x * blockDim.x + threadIdx.x;  // [0, 9216*1024/2)
    int frag = t >> 8;           // 256 dwords per fragment
    int r    = t & 255;
    int lane = r >> 3;
    int d    = r & 7;
    int kt   = frag >> 6;        // / NTILES
    int nt   = frag & (NTILES - 1);

    int n = nt * 16 + (lane & 15);
    int k = kt * 32 + (lane >> 4) * 8 + (d >> 2) * 16 + 2 * (d & 3);

    float lo, hi;
    if (k < IN_F) {              // base_weight^T : B[k,n] = base_weight[n,k]
        lo = base_w[(size_t)n * IN_F + k];
        hi = base_w[(size_t)n * IN_F + k + 1];
    } else {                     // spline_weight[n, (k-1024)] flattened (i*8+b)
        int kk = k - IN_F;
        lo = spline_w[(size_t)n * (IN_F * NB) + kk];
        hi = spline_w[(size_t)n * (IN_F * NB) + kk + 1];
    }
    Bp[t] = pack2(lo, hi);
}

// ---------------------------------------------------------------------------
// Kernel 3: bf16 WMMA GEMM.  Block = 4 waves (2x2) -> 128x128 tile; each wave
// owns 64x64 = 4x4 accumulator tiles (128 VGPRs of f32 accum).  Per K-tile:
// 16 b128 loads feeding 16 v_wmma_f32_16x16x32_bf16 (1.0 loads per WMMA);
// the compiler software-pipelines loads of tile kt+1 between the WMMAs of kt.
// ---------------------------------------------------------------------------
__global__ void __launch_bounds__(128, 1)
kan_gemm(const unsigned short* __restrict__ A,   // bf16 [4096 x 9216] row-major
         const unsigned* __restrict__ Bp,        // packed B fragments
         const float* __restrict__ bias,         // [1024]
         float* __restrict__ out) {              // f32 [4096 x 1024]
    const int lane   = threadIdx.x & 31;
    const int wave   = threadIdx.x >> 5;
    const int wm     = wave & 1;                 // 2 waves along M
    const int wn     = wave >> 1;                // 2 waves along N
    const int blockM = blockIdx.x * 128;
    const int blockN = blockIdx.y * 128;
    const int laneLo = lane & 15;
    const int laneHi = lane >> 4;

    v8f acc[4][4];
#pragma unroll
    for (int m = 0; m < 4; ++m)
#pragma unroll
        for (int j = 0; j < 4; ++j) acc[m][j] = (v8f){};

    // Single A base; M-subtile stride = 16*KDIM elements (294912 B, fits the
    // signed 24-bit VMEM immediate).  lane>=16 starts 8 halves in (K octet 1).
    const unsigned short* aBase =
        A + (size_t)(blockM + wm * 64 + laneLo) * KDIM + (size_t)laneHi * 8;

    // Single B base; N-subtile stride = 256 dwords (1 KiB immediate).
    const unsigned* bBase =
        Bp + (size_t)((blockN >> 4) + wn * 4) * 256 + (size_t)lane * 8;

    for (int kt = 0; kt < KTILES; ++kt) {
        union { uint4 u[2]; v16bf v; } af[4], bfr[4];

#pragma unroll
        for (int m = 0; m < 4; ++m) {
            const uint4* p = reinterpret_cast<const uint4*>(
                aBase + (size_t)m * (16 * KDIM) + kt * 32);
            af[m].u[0] = p[0];      // K octet 0 (or 8)
            af[m].u[1] = p[2];      // K octet 16 (or 24)
        }
#pragma unroll
        for (int j = 0; j < 4; ++j) {
            const uint4* p = reinterpret_cast<const uint4*>(
                bBase + (size_t)kt * (NTILES * 256) + j * 256);
            bfr[j].u[0] = p[0];
            bfr[j].u[1] = p[1];
        }

#pragma unroll
        for (int m = 0; m < 4; ++m)
#pragma unroll
            for (int j = 0; j < 4; ++j)
                acc[m][j] = __builtin_amdgcn_wmma_f32_16x16x32_bf16(
                    false, af[m].v, false, bfr[j].v, (short)0, acc[m][j],
                    false, false);
    }

    // Epilogue: C layout -> VGPR r: row = tileM + laneHi*8 + r, col = laneLo.
#pragma unroll
    for (int m = 0; m < 4; ++m) {
        int row0 = blockM + wm * 64 + m * 16 + laneHi * 8;
#pragma unroll
        for (int j = 0; j < 4; ++j) {
            int col = blockN + wn * 64 + j * 16 + laneLo;
            float bv = bias[col];
#pragma unroll
            for (int r = 0; r < 8; ++r)
                out[(size_t)(row0 + r) * OUT_F + col] = acc[m][j][r] + bv;
        }
    }
}

// ---------------------------------------------------------------------------
extern "C" void kernel_launch(void* const* d_in, const int* in_sizes, int n_in,
                              void* d_out, int out_size, void* d_ws, size_t ws_size,
                              hipStream_t stream) {
    const float* x        = (const float*)d_in[0];   // [4096,1024]
    const float* base_w   = (const float*)d_in[1];   // [1024,1024]
    const float* bias     = (const float*)d_in[2];   // [1024]
    const float* spline_w = (const float*)d_in[3];   // [1024,1024,8]
    float* out            = (float*)d_out;           // [4096,1024]

    const size_t A_bytes = (size_t)N_ROWS * KDIM * 2;          // 75,497,472
    unsigned short* A  = (unsigned short*)d_ws;
    unsigned*       Bp = (unsigned*)((char*)d_ws + A_bytes);   // 18,874,368

    // 1) A = [silu(x) | basis(x)] in bf16
    {
        int total = N_ROWS * IN_F;                    // 4,194,304
        prep_a<<<total / 256, 256, 0, stream>>>(x, A);
    }
    // 2) B pre-swizzled into WMMA fragment order
    {
        int total = (KDIM * OUT_F) / 2;               // 4,718,592 dwords
        prep_b<<<total / 256, 256, 0, stream>>>(base_w, spline_w, Bp);
    }
    // 3) GEMM + bias
    {
        dim3 grid(N_ROWS / 128, OUT_F / 128);         // (32, 8)
        kan_gemm<<<grid, 128, 0, stream>>>(A, Bp, bias, out);
    }
}
